// NeRFRenderer_87050397155455
// MI455X (gfx1250) — matrix operand order repaired and checked by reading, not compiled, not run on hardware
//
#include <hip/hip_runtime.h>
#include <hip/hip_bf16.h>
#include <math.h>

typedef __attribute__((ext_vector_type(16))) _Float16 v16h;
typedef __attribute__((ext_vector_type(8)))  float    v8f;

#define N_RAYS   4096
#define S_STEPS  768
#define HID      64
#define CHUNK    64
#define NCHUNK   (S_STEPS / CHUNK)
#define THREADS  128
#define NEARZ    0.01f
#define FARZ     0.81f

// A-fragment (16x32 f16) slot: lane = hi*16 + (m&15);
//   k_local bits: [4]=v>>2, [3]=hi, [2:1]=v&3, [0]=odd ; elem = 2v+odd
__device__ __forceinline__ int a_slot(int m_in_tile, int k_local) {
  int hi  = (k_local >> 3) & 1;
  int v   = ((k_local >> 4) << 2) | ((k_local >> 1) & 3);
  int odd = k_local & 1;
  int lane = hi * 16 + (m_in_tile & 15);
  return lane * 16 + (2 * v + odd);
}

__global__ __launch_bounds__(THREADS) void nvsf_render_kernel(
    const float* __restrict__ rays_o, const float* __restrict__ rays_d,
    const float* __restrict__ time_in,
    const float* __restrict__ W1,  const float* __restrict__ b1,
    const float* __restrict__ W2,  const float* __restrict__ b2,
    const float* __restrict__ w_sig, const float* __restrict__ b_sig,
    const float* __restrict__ Wc1, const float* __restrict__ bc1,
    const float* __restrict__ Wc2, const float* __restrict__ bc2,
    float* __restrict__ out_depth, float* __restrict__ out_image,
    float* __restrict__ out_wsum,  float* __restrict__ out_weights,
    float* __restrict__ out_zvals)
{
  // B-fragment pre-swizzled weights: 8 frags (nt:4 x ks:2) x 32 lanes x 16 halfs
  __shared__ __align__(32) _Float16 sW2f [8 * 32 * 16];
  __shared__ __align__(32) _Float16 sWc1f[8 * 32 * 16];
  // A-fragment activations: 8 frags (wave:4 x ks:2) x 32 lanes x 16 halfs
  __shared__ __align__(32) _Float16 sAf  [8 * 32 * 16];   // H1 (frag layout)
  __shared__ __align__(32) _Float16 sH2f [8 * 32 * 16];   // H2 (frag layout)
  __shared__ float sW1[4 * HID];
  __shared__ float sb1[HID], sb2[HID], swsig[HID], sc0[HID];
  __shared__ float sWc2[HID * 2];
  __shared__ float sSig[CHUNK];
  __shared__ float sRgb[CHUNK * 2];
  __shared__ float sWgt[CHUNK];

  const int tid  = threadIdx.x;
  const int ray  = blockIdx.x;
  const int wave = tid >> 5;
  const int lane = tid & 31;
  const int lm   = lane & 15;
  const int hi   = lane >> 4;

  // hint weight tensors into cache early (global_prefetch_b8)
  __builtin_prefetch(W2  + (tid << 3), 0, 3);
  __builtin_prefetch(Wc1 + (tid << 3), 0, 3);

  // ---- per-ray constants ----
  const float ox = rays_o[ray * 3 + 0], oy = rays_o[ray * 3 + 1], oz = rays_o[ray * 3 + 2];
  const float dx = rays_d[ray * 3 + 0], dy = rays_d[ray * 3 + 1], dz = rays_d[ray * 3 + 2];
  const float tv = time_in[ray];
  const float bs   = b_sig[0];
  const float bc20 = bc2[0], bc21 = bc2[1];

  const float zstep = (FARZ - NEARZ) / (float)(S_STEPS - 1);   // linspace spacing
  const float dlast = (FARZ - NEARZ) / (float)S_STEPS;         // sample_dist

  // ---- load small params ----
  for (int i = tid; i < 4 * HID; i += THREADS) sW1[i] = W1[i];
  for (int i = tid; i < HID; i += THREADS) {
    sb1[i]   = b1[i];
    sb2[i]   = b2[i];
    swsig[i] = w_sig[i];
    sWc2[i * 2 + 0] = Wc2[i * 2 + 0];
    sWc2[i * 2 + 1] = Wc2[i * 2 + 1];
    // dirs part of color head is constant along the ray: fold into bias
    sc0[i] = bc1[i] + dx * Wc1[0 * HID + i] + dy * Wc1[1 * HID + i] + dz * Wc1[2 * HID + i];
  }
  // ---- pre-swizzle W2 / Wc1[3:,:] into B-fragment layout ----
  for (int fi = tid; fi < 8 * 32; fi += THREADS) {   // one lane-slot (16 halfs) per iter
    int frag = fi >> 5, ln = fi & 31;
    int nt = frag >> 1, ks = frag & 1;
    int n  = nt * 16 + (ln & 15);
    int bh = ln >> 4;
#pragma unroll
    for (int e = 0; e < 16; ++e) {
      int k = ks * 32 + ((e >> 3) << 4) + bh * 8 + (e & 7);   // B layout: k_local = (e>>3)*16 + hi*8 + (e&7)
      sW2f [fi * 16 + e] = (_Float16)W2 [k * HID + n];
      sWc1f[fi * 16 + e] = (_Float16)Wc1[(3 + k) * HID + n];
    }
  }
  __syncthreads();

  // compositing carries (wave 0 lanes hold partial sums; Tcar uniform in wave 0)
  float Tcar = 1.0f, dep = 0.0f, im0 = 0.0f, im1 = 0.0f, wsum = 0.0f;

  for (int c = 0; c < NCHUNK; ++c) {
    const int base = c * CHUNK;

    // ---- stage 1: H1 = relu([x,t] @ W1 + b1), stored in A-fragment layout ----
    for (int idx = tid; idx < CHUNK * HID; idx += THREADS) {
      int sl = idx >> 6, h = idx & 63;
      float z  = NEARZ + zstep * (float)(base + sl);
      float px = fminf(fmaxf(ox + dx * z, -1.0f), 1.0f);
      float py = fminf(fmaxf(oy + dy * z, -1.0f), 1.0f);
      float pz = fminf(fmaxf(oz + dz * z, -1.0f), 1.0f);
      float v = px * sW1[0 * HID + h] + py * sW1[1 * HID + h] +
                pz * sW1[2 * HID + h] + tv * sW1[3 * HID + h] + sb1[h];
      v = fmaxf(v, 0.0f);
      int w2 = sl >> 4, ks = h >> 5, kl = h & 31;
      sAf[((w2 * 2 + ks) * 32) * 16 + a_slot(sl, kl)] = (_Float16)v;
    }
    __syncthreads();

    // ---- stage 2: H2 = relu(H1 @ W2 + b2) [WMMA]; sigma partials in registers ----
    float sig_part[8];
#pragma unroll
    for (int r = 0; r < 8; ++r) sig_part[r] = 0.0f;

#pragma unroll
    for (int nt = 0; nt < 4; ++nt) {
      v8f acc = {};
#pragma unroll
      for (int ks = 0; ks < 2; ++ks) {
        v16h a = *(const v16h*)&sAf [((wave * 2 + ks) * 32 + lane) * 16];
        v16h b = *(const v16h*)&sW2f[((nt   * 2 + ks) * 32 + lane) * 16];
        acc = __builtin_amdgcn_wmma_f32_16x16x32_f16(false, a, false, b,
                                                     (short)0, acc, false, false);
      }
      const int n    = nt * 16 + lm;
      const float wsg = swsig[n];
#pragma unroll
      for (int r = 0; r < 8; ++r) {
        int m = wave * 16 + r + 8 * hi;
        float v = fmaxf(acc[r] + sb2[n], 0.0f);
        sig_part[r] += v * wsg;                    // sigma head partial (this lane's column)
        int w2 = m >> 4, ks2 = n >> 5, kl = n & 31;
        sH2f[((w2 * 2 + ks2) * 32) * 16 + a_slot(m, kl)] = (_Float16)v;
      }
    }
    // reduce sigma partials across the 16 lanes sharing a row (xor stays in hi-group)
#pragma unroll
    for (int r = 0; r < 8; ++r) {
#pragma unroll
      for (int mask = 8; mask >= 1; mask >>= 1)
        sig_part[r] += __shfl_xor(sig_part[r], mask, 32);
    }
    if (lm == 0) {
#pragma unroll
      for (int r = 0; r < 8; ++r) {
        int m = wave * 16 + r + 8 * hi;
        float s = sig_part[r] + bs;
        sSig[m] = (s > 20.0f) ? s : log1pf(__expf(s));   // softplus
      }
    }
    __syncthreads();

    // ---- stage 3: C1 = relu(H2 @ Wc1[3:,:] + c0) [WMMA]; rgb partials in registers ----
    float rgb0[8], rgb1[8];
#pragma unroll
    for (int r = 0; r < 8; ++r) { rgb0[r] = 0.0f; rgb1[r] = 0.0f; }

#pragma unroll
    for (int nt = 0; nt < 4; ++nt) {
      v8f acc = {};
#pragma unroll
      for (int ks = 0; ks < 2; ++ks) {
        v16h a = *(const v16h*)&sH2f [((wave * 2 + ks) * 32 + lane) * 16];
        v16h b = *(const v16h*)&sWc1f[((nt   * 2 + ks) * 32 + lane) * 16];
        acc = __builtin_amdgcn_wmma_f32_16x16x32_f16(false, a, false, b,
                                                     (short)0, acc, false, false);
      }
      const int n     = nt * 16 + lm;
      const float wc0 = sWc2[n * 2 + 0];
      const float wc1 = sWc2[n * 2 + 1];
#pragma unroll
      for (int r = 0; r < 8; ++r) {
        float v = fmaxf(acc[r] + sc0[n], 0.0f);
        rgb0[r] += v * wc0;
        rgb1[r] += v * wc1;
      }
    }
#pragma unroll
    for (int r = 0; r < 8; ++r) {
#pragma unroll
      for (int mask = 8; mask >= 1; mask >>= 1) {
        rgb0[r] += __shfl_xor(rgb0[r], mask, 32);
        rgb1[r] += __shfl_xor(rgb1[r], mask, 32);
      }
    }
    if (lm == 0) {
#pragma unroll
      for (int r = 0; r < 8; ++r) {
        int m = wave * 16 + r + 8 * hi;
        sRgb[m * 2 + 0] = 1.0f / (1.0f + __expf(-(rgb0[r] + bc20)));
        sRgb[m * 2 + 1] = 1.0f / (1.0f + __expf(-(rgb1[r] + bc21)));
      }
    }
    __syncthreads();

    // ---- alpha compositing: wave-parallel prefix product scan (wave 0) ----
    if (wave == 0) {
#pragma unroll
      for (int half = 0; half < 2; ++half) {
        int   i     = half * 32 + lane;
        int   sg    = base + i;
        float z     = NEARZ + zstep * (float)sg;
        float delta = (sg == S_STEPS - 1) ? dlast : zstep;
        float alpha = 1.0f - __expf(-delta * sSig[i]);
        float g     = 1.0f - alpha + 1e-15f;
        // inclusive prefix product of g across the wave
        float p = g;
#pragma unroll
        for (int st = 1; st < 32; st <<= 1) {
          float o = __shfl_up(p, st, 32);
          if (lane >= st) p *= o;
        }
        float excl = __shfl_up(p, 1, 32);
        if (lane == 0) excl = 1.0f;
        float w = alpha * Tcar * excl;             // weight_i = alpha_i * T_i
        sWgt[i] = w;
        dep  += w * z;
        im0  += w * sRgb[i * 2 + 0];
        im1  += w * sRgb[i * 2 + 1];
        wsum += w;
        Tcar *= __shfl(p, 31, 32);                 // carry transmittance
      }
    }
    __syncthreads();

    // ---- stream out weights / z_vals for this chunk ----
    if (tid < CHUNK) {
      int sg = base + tid;
      out_weights[ray * S_STEPS + sg] = sWgt[tid];
      out_zvals  [ray * S_STEPS + sg] = NEARZ + zstep * (float)sg;
    }
    __syncthreads();
  }

  // ---- final reduction of per-lane composites (wave 0) ----
  if (wave == 0) {
#pragma unroll
    for (int mask = 16; mask >= 1; mask >>= 1) {
      dep  += __shfl_xor(dep,  mask, 32);
      im0  += __shfl_xor(im0,  mask, 32);
      im1  += __shfl_xor(im1,  mask, 32);
      wsum += __shfl_xor(wsum, mask, 32);
    }
    if (lane == 0) {
      out_depth[ray]         = dep;
      out_image[ray * 2 + 0] = im0;
      out_image[ray * 2 + 1] = im1;
      out_wsum[ray]          = wsum;
    }
  }
}

extern "C" void kernel_launch(void* const* d_in, const int* in_sizes, int n_in,
                              void* d_out, int out_size, void* d_ws, size_t ws_size,
                              hipStream_t stream) {
  const float* rays_o  = (const float*)d_in[0];
  const float* rays_d  = (const float*)d_in[1];
  const float* time_in = (const float*)d_in[2];
  const float* W1      = (const float*)d_in[3];
  const float* b1      = (const float*)d_in[4];
  const float* W2      = (const float*)d_in[5];
  const float* b2      = (const float*)d_in[6];
  const float* w_sig   = (const float*)d_in[7];
  const float* b_sig   = (const float*)d_in[8];
  const float* Wc1     = (const float*)d_in[9];
  const float* bc1     = (const float*)d_in[10];
  const float* Wc2     = (const float*)d_in[11];
  const float* bc2     = (const float*)d_in[12];

  float* out = (float*)d_out;
  float* out_depth   = out;                                   // [4096]
  float* out_image   = out + N_RAYS;                          // [4096,2]
  float* out_wsum    = out + N_RAYS * 3;                      // [4096]
  float* out_weights = out + N_RAYS * 4;                      // [4096,768]
  float* out_zvals   = out_weights + N_RAYS * S_STEPS;        // [4096,768]

  nvsf_render_kernel<<<dim3(N_RAYS), dim3(THREADS), 0, stream>>>(
      rays_o, rays_d, time_in, W1, b1, W2, b2, w_sig, b_sig,
      Wc1, bc1, Wc2, bc2,
      out_depth, out_image, out_wsum, out_weights, out_zvals);
}
